// HybridLongSequenceBackbone_46772193853555
// MI455X (gfx1250) — matrix operand order repaired and checked by reading, not compiled, not run on hardware
//
#include <hip/hip_runtime.h>
#include <cmath>

// ---------------------------------------------------------------------------
// Types for CDNA5 WMMA (wave32, 16x16x32 f16 -> f32 accumulate)
// ---------------------------------------------------------------------------
typedef _Float16 h16_t;
typedef __attribute__((ext_vector_type(16))) _Float16 v16h;
typedef __attribute__((ext_vector_type(8)))  _Float16 v8h;
typedef __attribute__((ext_vector_type(8)))  float    v8f;

#define LP 40   // LDS row pitch in halves (32 data + 8 pad -> conflict-free b128 reads)

__device__ __forceinline__ float gelu_exact(float v) {
    return 0.5f * v * (1.0f + erff(v * 0.70710678118654752f));
}
__device__ __forceinline__ float sigmoid_f(float v) {
    return 1.0f / (1.0f + expf(-v));
}

// ---------------------------------------------------------------------------
// Generic WMMA GEMM: out[M,N] = act(A[M,K] @ W[K,N] + bias) (+ res)
// A: f16 row-major (lda halves). WT: f16 [N][K] (K contiguous).
// Block tile 128M x 64N, 8 waves; each wave: 16M x 64N (4 WMMAs / K-step).
// ---------------------------------------------------------------------------
__global__ __launch_bounds__(256)
void gemm_wmma_kernel(const h16_t* __restrict__ A, int lda,
                      const h16_t* __restrict__ WT,
                      const float* __restrict__ bias,
                      const float* __restrict__ res, int ldres,
                      float* __restrict__ outF, int ldo,
                      h16_t* __restrict__ outH, int ldoh,
                      int K, int act)
{
    __shared__ __align__(16) h16_t lA[128 * LP];
    __shared__ __align__(16) h16_t lB[64 * LP];

    const int tid  = threadIdx.x;
    const int wave = tid >> 5;
    const int lane = tid & 31;
    const int m0 = blockIdx.x * 128;
    const int n0 = blockIdx.y * 64;

    const int arow = tid >> 1;            // 0..127
    const int acol = (tid & 1) * 16;      // half offset 0/16
    const int bn   = tid >> 2;            // 0..63
    const int bk   = (tid & 3) * 8;       // 0..24

    const int lhalf = lane & 15;
    const int hiHalf = lane >> 4;         // 0: lanes 0-15, 1: lanes 16-31

    v8f acc[4] = {v8f{}, v8f{}, v8f{}, v8f{}};

    for (int k0 = 0; k0 < K; k0 += 32) {
        // Stage A tile 128x32 (f16, row-major) into LDS
        const h16_t* ag = A + (size_t)(m0 + arow) * lda + (k0 + acol);
        *(v8h*)&lA[arow * LP + acol]     = *(const v8h*)(ag);
        *(v8h*)&lA[arow * LP + acol + 8] = *(const v8h*)(ag + 8);
        // Stage B tile 64(N)x32(K) into LDS as [n][k]
        const h16_t* bg = WT + (size_t)(n0 + bn) * K + (k0 + bk);
        *(v8h*)&lB[bn * LP + bk] = *(const v8h*)(bg);
        if (k0 + 32 < K) {                         // prefetch next K tile
            __builtin_prefetch(ag + 32, 0, 1);     // -> global_prefetch_b8
            __builtin_prefetch(bg + 32, 0, 1);
        }
        __syncthreads();

        // A fragment per ISA layout: lanes0-15 row=lhalf, K {0..7,16..23};
        // lanes16-31 row=lhalf, K {8..15,24..31}
        const int rowA = wave * 16 + lhalf;
        const int ka   = hiHalf * 8;
        v8h alo = *(const v8h*)&lA[rowA * LP + ka];
        v8h ahi = *(const v8h*)&lA[rowA * LP + ka + 16];
        v16h afrag;
#pragma unroll
        for (int t = 0; t < 8; ++t) { afrag[t] = alo[t]; afrag[t + 8] = ahi[t]; }

#pragma unroll
        for (int j = 0; j < 4; ++j) {
            // B fragment: lanes0-15 N=lhalf K {0..7,16..23}; lanes16-31 N=lhalf K {8..15,24..31}
            const int nloc = j * 16 + lhalf;
            const int kb = hiHalf * 8;
            v8h blo = *(const v8h*)&lB[nloc * LP + kb];
            v8h bhi = *(const v8h*)&lB[nloc * LP + kb + 16];
            v16h bfrag;
#pragma unroll
            for (int t = 0; t < 8; ++t) { bfrag[t] = blo[t]; bfrag[t + 8] = bhi[t]; }
            acc[j] = __builtin_amdgcn_wmma_f32_16x16x32_f16(
                false, afrag, false, bfrag, (short)0, acc[j], false, false);
        }
        __syncthreads();
    }

    // C/D layout: VGPR r <-> M = r + 8*hiHalf, N = lane%16
    const int mbase = m0 + wave * 16 + hiHalf * 8;
#pragma unroll
    for (int j = 0; j < 4; ++j) {
        const int col = n0 + j * 16 + lhalf;
        const float bv = bias ? bias[col] : 0.0f;
#pragma unroll
        for (int r = 0; r < 8; ++r) {
            const int row = mbase + r;
            float v = acc[j][r] + bv;
            if (act == 1) v = gelu_exact(v);
            if (res)  v += res[(size_t)row * ldres + col];
            if (outF) outF[(size_t)row * ldo + col] = v;
            if (outH) outH[(size_t)row * ldoh + col] = (h16_t)v;
        }
    }
}

// ---------------------------------------------------------------------------
// Weight convert + transpose: f32 [K][N] -> f16 [N][K]
// ---------------------------------------------------------------------------
__global__ void convT_kernel(const float* __restrict__ src, h16_t* __restrict__ dst,
                             int K, int N)
{
    const int e = blockIdx.x * 256 + threadIdx.x;
    if (e >= K * N) return;
    const int k = e / N, n = e - k * N;
    dst[(size_t)n * K + k] = (h16_t)src[e];
}

// ---------------------------------------------------------------------------
// cond projections: condp[kind][B][256], kind: 0=film_g 1=film_b 2=agn_w 3=agn_b
// ---------------------------------------------------------------------------
__global__ void cond_proj_kernel(const float* __restrict__ cond,
    const float* __restrict__ fgw, const float* __restrict__ fgb,
    const float* __restrict__ fbw, const float* __restrict__ fbb,
    const float* __restrict__ aww, const float* __restrict__ awb,
    const float* __restrict__ abw, const float* __restrict__ abb,
    float* __restrict__ condp)
{
    const int t = blockIdx.x * 256 + threadIdx.x;
    if (t >= 1024) return;
    const int b = t >> 8, j = t & 255;
    float s0 = fgb[j], s1 = fbb[j], s2 = awb[j], s3 = abb[j];
    for (int k = 0; k < 32; ++k) {
        const float cv = cond[b * 32 + k];
        s0 += cv * fgw[k * 256 + j];
        s1 += cv * fbw[k * 256 + j];
        s2 += cv * aww[k * 256 + j];
        s3 += cv * abw[k * 256 + j];
    }
    condp[(0 * 4 + b) * 256 + j] = s0;
    condp[(1 * 4 + b) * 256 + j] = s1;
    condp[(2 * 4 + b) * 256 + j] = s2;
    condp[(3 * 4 + b) * 256 + j] = s3;
}

// ---------------------------------------------------------------------------
// LN + t_emb + FiLM  (one block per token row)
// ---------------------------------------------------------------------------
__global__ __launch_bounds__(256)
void ln_film_kernel(const float* __restrict__ x,
                    const float* __restrict__ g, const float* __restrict__ bl,
                    const float* __restrict__ temb, float ts,
                    const float* __restrict__ condp,
                    float* __restrict__ hbuf)
{
    const int row = blockIdx.x;
    const int b = row >> 12;       // L = 4096
    const int j = threadIdx.x;
    __shared__ float red[256];
    const float xv = x[(size_t)row * 256 + j];
    red[j] = xv; __syncthreads();
    for (int s = 128; s > 0; s >>= 1) { if (j < s) red[j] += red[j + s]; __syncthreads(); }
    const float mean = red[0] * (1.0f / 256.0f); __syncthreads();
    const float d = xv - mean;
    red[j] = d * d; __syncthreads();
    for (int s = 128; s > 0; s >>= 1) { if (j < s) red[j] += red[j + s]; __syncthreads(); }
    const float var = red[0] * (1.0f / 256.0f);
    const float ln = d * rsqrtf(var + 1e-5f) * g[j] + bl[j] + ts * temb[b * 256 + j];
    const float fg = condp[(0 * 4 + b) * 256 + j];
    const float fb = condp[(1 * 4 + b) * 256 + j];
    hbuf[(size_t)row * 256 + j] = ln * (1.0f + fg) + fb;
}

// Deterministic GroupNorm stats: one block per (b,g); reduce L*8 = 32768 elems
__global__ __launch_bounds__(256)
void group_stats_kernel(const float* __restrict__ h, float* __restrict__ stats)
{
    const int bg = blockIdx.x;        // b*32+g
    const int b = bg >> 5, g = bg & 31;
    const int t = threadIdx.x;
    __shared__ float s1[256], s2[256];
    float a = 0.f, q = 0.f;
    for (int e = t; e < 4096 * 8; e += 256) {
        const int l = e >> 3, c = e & 7;
        const float v = h[((size_t)(b * 4096 + l)) * 256 + g * 8 + c];
        a += v; q += v * v;
    }
    s1[t] = a; s2[t] = q; __syncthreads();
    for (int s = 128; s > 0; s >>= 1) {
        if (t < s) { s1[t] += s1[t + s]; s2[t] += s2[t + s]; }
        __syncthreads();
    }
    if (t == 0) { stats[bg * 2] = s1[0]; stats[bg * 2 + 1] = s2[0]; }
}

// Apply GN + AGN; write f32 and f16 (f16 feeds the next GEMM's A)
__global__ void gn_apply_kernel(const float* __restrict__ hbuf,
                                const float* __restrict__ stats,
                                const float* __restrict__ condp,
                                float* __restrict__ hout, h16_t* __restrict__ h16)
{
    const int idx = blockIdx.x * 256 + threadIdx.x;
    const int j = idx & 255;
    const int row = idx >> 8;
    const int b = row >> 12;
    const int gi = b * 32 + (j >> 3);
    const float inv_n = 1.0f / (4096.0f * 8.0f);
    const float m = stats[gi * 2] * inv_n;
    const float v = stats[gi * 2 + 1] * inv_n - m * m;
    const float aw = condp[(2 * 4 + b) * 256 + j];
    const float ab = condp[(3 * 4 + b) * 256 + j];
    const float h = (hbuf[idx] - m) * rsqrtf(v + 1e-5f) * (1.0f + aw) + ab;
    hout[idx] = h;
    h16[idx] = (h16_t)h;
}

// Mamba causal depthwise conv (DC=4, pad (3,0)) + SiLU.  u = ug[:, :512]
__global__ void mamba_conv_kernel(const float* __restrict__ ug,
                                  const float* __restrict__ w,
                                  const float* __restrict__ bias,
                                  float* __restrict__ uc, h16_t* __restrict__ uc16)
{
    const int idx = blockIdx.x * 256 + threadIdx.x;   // < M*512 exact
    const int c = idx & 511;
    const int row = idx >> 9;
    const int l = row & 4095;
    const float* up = ug + (size_t)row * 1024 + c;
    float acc = bias[c];
#pragma unroll
    for (int k = 0; k < 4; ++k) {
        const int ls = l - 3 + k;
        if (ls >= 0) acc += w[c * 4 + k] * up[(k - 3) * 1024];
    }
    const float s = acc * sigmoid_f(acc);
    uc[idx] = s;
    uc16[idx] = (h16_t)s;
}

// dt = clip(softplus(dtraw)); la = -exp(A_log)*dt (in-place); bb = Bt*dt (in-place)
__global__ void dt_prep_kernel(float* __restrict__ dtb, float* __restrict__ btb,
                               const float* __restrict__ alog)
{
    const int idx = blockIdx.x * 256 + threadIdx.x;   // < M*64 exact
    const int s = idx & 63;
    const float raw = dtb[idx];
    const float sp = (raw > 20.0f) ? raw : log1pf(expf(raw));
    const float dt = fminf(fmaxf(sp, 0.001f), 0.1f);
    dtb[idx] = -expf(alog[s]) * dt;
    btb[idx] *= dt;
}

// Selective-scan: h_t = exp(la_t)*h_{t-1} + bb_t per (b,s); output hC16 = h*Ct.
// Chunked (16/thread) + Hillis-Steele segment-composition scan across 256 threads.
__global__ __launch_bounds__(256)
void scan_kernel(const float* __restrict__ la, const float* __restrict__ bb,
                 const float* __restrict__ Ct, h16_t* __restrict__ hC16)
{
    const int bs = blockIdx.x;             // b*64 + s
    const int b = bs >> 6, s = bs & 63;
    const int t = threadIdx.x;
    __shared__ float sA[256], sB[256];
    const size_t base = ((size_t)b * 4096) * 64 + s;
    const int l0 = t * 16;

    float Ac = 1.0f, Bc = 0.0f;
    for (int u = 0; u < 16; ++u) {
        const size_t id = base + (size_t)(l0 + u) * 64;
        const float a = expf(la[id]);
        Bc = a * Bc + bb[id];
        Ac = a * Ac;
    }
    sA[t] = Ac; sB[t] = Bc; __syncthreads();
    for (int off = 1; off < 256; off <<= 1) {
        float pA = 1.0f, pB = 0.0f;
        if (t >= off) { pA = sA[t - off]; pB = sB[t - off]; }
        __syncthreads();
        if (t >= off) {
            const float nA = sA[t] * pA;
            const float nB = sA[t] * pB + sB[t];
            sA[t] = nA; sB[t] = nB;
        }
        __syncthreads();
    }
    float h = (t == 0) ? 0.0f : sB[t - 1];
    for (int u = 0; u < 16; ++u) {
        const size_t id = base + (size_t)(l0 + u) * 64;
        const float a = expf(la[id]);
        h = a * h + bb[id];
        hC16[id] = (h16_t)(h * Ct[id]);
    }
}

// y = (ydi + uc*D) * sigmoid(gate);  gate = ug[:, 512:1024]
__global__ void gate_kernel(const float* __restrict__ ydi,
                            const float* __restrict__ uc,
                            const float* __restrict__ ug,
                            const float* __restrict__ Dv,
                            h16_t* __restrict__ y16)
{
    const int idx = blockIdx.x * 256 + threadIdx.x;   // < M*512 exact
    const int c = idx & 511;
    const int row = idx >> 9;
    const float sg = sigmoid_f(ug[(size_t)row * 1024 + 512 + c]);
    y16[idx] = (h16_t)((ydi[idx] + uc[idx] * Dv[c]) * sg);
}

// Multi-scale dilated depthwise conv (K=3, 'same') + exact GELU -> f16
__global__ void ms_conv_kernel(const float* __restrict__ h,
                               const float* __restrict__ w,
                               const float* __restrict__ bias,
                               int dil, h16_t* __restrict__ out16)
{
    const int idx = blockIdx.x * 256 + threadIdx.x;   // < M*256 exact
    const int c = idx & 255;
    const int row = idx >> 8;
    const int l = row & 4095;
    const float* hp = h + (size_t)row * 256 + c;
    float acc = bias[c];
#pragma unroll
    for (int k = 0; k < 3; ++k) {
        const int ls = l + (k - 1) * dil;
        if (ls >= 0 && ls < 4096) acc += w[c * 3 + k] * hp[(k - 1) * dil * 256];
    }
    out16[idx] = (h16_t)gelu_exact(acc);
}

// Final LayerNorm -> d_out
__global__ __launch_bounds__(256)
void final_ln_kernel(const float* __restrict__ x, const float* __restrict__ g,
                     const float* __restrict__ bl, float* __restrict__ out)
{
    const int row = blockIdx.x;
    const int j = threadIdx.x;
    __shared__ float red[256];
    const float xv = x[(size_t)row * 256 + j];
    red[j] = xv; __syncthreads();
    for (int s = 128; s > 0; s >>= 1) { if (j < s) red[j] += red[j + s]; __syncthreads(); }
    const float mean = red[0] * (1.0f / 256.0f); __syncthreads();
    const float d = xv - mean;
    red[j] = d * d; __syncthreads();
    for (int s = 128; s > 0; s >>= 1) { if (j < s) red[j] += red[j + s]; __syncthreads(); }
    const float var = red[0] * (1.0f / 256.0f);
    out[(size_t)row * 256 + j] = d * rsqrtf(var + 1e-5f) * g[j] + bl[j];
}

// ---------------------------------------------------------------------------
// Host orchestration
// ---------------------------------------------------------------------------
extern "C" void kernel_launch(void* const* d_in, const int* in_sizes, int n_in,
                              void* d_out, int out_size, void* d_ws, size_t ws_size,
                              hipStream_t stream)
{
    (void)in_sizes; (void)n_in; (void)out_size; (void)ws_size;
    const int NL = 4, Dm = 256, DSd = 64, DId = 512, CDm = 32;
    const int Bq = 4, Lq = 4096;
    const int M = Bq * Lq;                  // 16384 token rows

    // --- inputs (setup_inputs dict order: x, t_emb, cond, then params) ---
    const float* x_in     = (const float*)d_in[0];
    const float* t_emb    = (const float*)d_in[1];
    const float* cond     = (const float*)d_in[2];
    const float* ln1_g    = (const float*)d_in[3];
    const float* ln1_b    = (const float*)d_in[4];
    const float* ln2_g    = (const float*)d_in[5];
    const float* ln2_b    = (const float*)d_in[6];
    const float* ln3_g    = (const float*)d_in[7];
    const float* ln3_b    = (const float*)d_in[8];
    const float* final_g  = (const float*)d_in[9];
    const float* final_b  = (const float*)d_in[10];
    const float* in_w     = (const float*)d_in[11];
    const float* in_b     = (const float*)d_in[12];
    const float* conv_w   = (const float*)d_in[13];
    const float* conv_b   = (const float*)d_in[14];
    const float* B_w      = (const float*)d_in[15];
    const float* C_w      = (const float*)d_in[16];
    const float* dt_w     = (const float*)d_in[17];
    const float* dt_b     = (const float*)d_in[18];
    const float* A_log    = (const float*)d_in[19];
    const float* D_vec    = (const float*)d_in[20];
    const float* os_w     = (const float*)d_in[21];
    const float* out_w    = (const float*)d_in[22];
    const float* out_b    = (const float*)d_in[23];
    const float* dw_w     = (const float*)d_in[24];
    const float* dw_b     = (const float*)d_in[25];
    const float* pw_w     = (const float*)d_in[26];
    const float* pw_b     = (const float*)d_in[27];
    const float* f1_w     = (const float*)d_in[28];
    const float* f1_b     = (const float*)d_in[29];
    const float* f2_w     = (const float*)d_in[30];
    const float* f2_b     = (const float*)d_in[31];
    const float* ffn1_w   = (const float*)d_in[32];
    const float* ffn1_b   = (const float*)d_in[33];
    const float* ffn2_w   = (const float*)d_in[34];
    const float* ffn2_b   = (const float*)d_in[35];
    const float* fg_w     = (const float*)d_in[36];
    const float* fg_b     = (const float*)d_in[37];
    const float* fb_w     = (const float*)d_in[38];
    const float* fb_b     = (const float*)d_in[39];
    const float* aw_w     = (const float*)d_in[40];
    const float* aw_b     = (const float*)d_in[41];
    const float* ab_w     = (const float*)d_in[42];
    const float* ab_b     = (const float*)d_in[43];

    // --- workspace layout ---
    char* base = (char*)d_ws;
    size_t off = 0;
    auto take = [&](size_t bytes) -> char* {
        char* p = base + off;
        off += (bytes + 255) & ~(size_t)255;
        return p;
    };
    float* xb    = (float*) take((size_t)M * 256 * 4);
    float* hb    = (float*) take((size_t)M * 256 * 4);
    h16_t* h16b  = (h16_t*) take((size_t)M * 256 * 2);
    float* ug    = (float*) take((size_t)M * 1024 * 4);   // alias: ff16
    float* ucb   = (float*) take((size_t)M * 512 * 4);    // alias: concat16
    h16_t* uc16  = (h16_t*) take((size_t)M * 512 * 2);    // alias: mid16
    float* dtb   = (float*) take((size_t)M * 64 * 4);
    float* btb   = (float*) take((size_t)M * 64 * 4);
    float* ctb   = (float*) take((size_t)M * 64 * 4);
    h16_t* hC16  = (h16_t*) take((size_t)M * 64 * 2);
    float* ydi   = (float*) take((size_t)M * 512 * 4);
    h16_t* y16   = (h16_t*) take((size_t)M * 512 * 2);    // alias: c16
    float* stats = (float*) take(256 * 4);
    float* condp = (float*) take(4096 * 4);
    h16_t* inT   = (h16_t*) take((size_t)NL * 1024 * 256 * 2);
    h16_t* dtT   = (h16_t*) take((size_t)NL * 64 * 512 * 2);
    h16_t* BT    = (h16_t*) take((size_t)NL * 64 * 512 * 2);
    h16_t* CT    = (h16_t*) take((size_t)NL * 64 * 512 * 2);
    h16_t* osT   = (h16_t*) take((size_t)NL * 512 * 64 * 2);
    h16_t* outT  = (h16_t*) take((size_t)NL * 256 * 512 * 2);
    h16_t* pwT   = (h16_t*) take((size_t)NL * 3 * 256 * 256 * 2);
    h16_t* f1T   = (h16_t*) take((size_t)NL * 512 * 768 * 2);
    h16_t* f2T   = (h16_t*) take((size_t)NL * 256 * 512 * 2);
    h16_t* ffn1T = (h16_t*) take((size_t)NL * 1024 * 256 * 2);
    h16_t* ffn2T = (h16_t*) take((size_t)NL * 256 * 1024 * 2);
    h16_t* concat16 = (h16_t*)ucb;    // 24 MB <= 32 MB
    h16_t* mid16    = uc16;           // 16 MB
    h16_t* ff16     = (h16_t*)ug;     // 32 MB <= 64 MB
    h16_t* c16      = y16;            //  8 MB <= 16 MB

    (void)hipMemcpyAsync(xb, x_in, (size_t)M * 256 * 4, hipMemcpyDeviceToDevice, stream);

    auto convT = [&](const float* src, h16_t* dst, int K, int N) {
        const int tot = K * N;
        convT_kernel<<<(tot + 255) / 256, 256, 0, stream>>>(src, dst, K, N);
    };
    // one-time f32 -> f16 [N][K] weight conversion for all layers
    for (int i = 0; i < NL; ++i) {
        convT(in_w   + (size_t)i * 256 * 1024, inT   + (size_t)i * 1024 * 256, 256, 1024);
        convT(dt_w   + (size_t)i * 512 * 64,   dtT   + (size_t)i * 64 * 512,   512, 64);
        convT(B_w    + (size_t)i * 512 * 64,   BT    + (size_t)i * 64 * 512,   512, 64);
        convT(C_w    + (size_t)i * 512 * 64,   CT    + (size_t)i * 64 * 512,   512, 64);
        convT(os_w   + (size_t)i * 64 * 512,   osT   + (size_t)i * 512 * 64,   64, 512);
        convT(out_w  + (size_t)i * 512 * 256,  outT  + (size_t)i * 256 * 512,  512, 256);
        for (int j = 0; j < 3; ++j)
            convT(pw_w + ((size_t)i * 3 + j) * 256 * 256,
                  pwT  + ((size_t)i * 3 + j) * 256 * 256, 256, 256);
        convT(f1_w   + (size_t)i * 768 * 512,  f1T   + (size_t)i * 512 * 768,  768, 512);
        convT(f2_w   + (size_t)i * 512 * 256,  f2T   + (size_t)i * 256 * 512,  512, 256);
        convT(ffn1_w + (size_t)i * 256 * 1024, ffn1T + (size_t)i * 1024 * 256, 256, 1024);
        convT(ffn2_w + (size_t)i * 1024 * 256, ffn2T + (size_t)i * 256 * 1024, 1024, 256);
    }

    auto gemm = [&](const h16_t* A, int lda, const h16_t* WT, const float* bias,
                    const float* res, int ldres, float* outF, int ldo,
                    h16_t* outH, int ldoh, int N, int K, int act) {
        dim3 grid(M / 128, N / 64);
        gemm_wmma_kernel<<<grid, 256, 0, stream>>>(A, lda, WT, bias, res, ldres,
                                                   outF, ldo, outH, ldoh, K, act);
    };
    auto modulate = [&](const float* lng, const float* lnb, float ts) {
        ln_film_kernel<<<M, 256, 0, stream>>>(xb, lng, lnb, t_emb, ts, condp, hb);
        group_stats_kernel<<<128, 256, 0, stream>>>(hb, stats);
        gn_apply_kernel<<<M, 256, 0, stream>>>(hb, stats, condp, hb, h16b);
    };

    const int dils[3] = {1, 4, 16};

    for (int i = 0; i < NL; ++i) {
        cond_proj_kernel<<<4, 256, 0, stream>>>(cond,
            fg_w + (size_t)i * CDm * Dm, fg_b + (size_t)i * Dm,
            fb_w + (size_t)i * CDm * Dm, fb_b + (size_t)i * Dm,
            aw_w + (size_t)i * CDm * Dm, aw_b + (size_t)i * Dm,
            ab_w + (size_t)i * CDm * Dm, ab_b + (size_t)i * Dm, condp);

        // ---- Mamba branch ----
        modulate(ln1_g + (size_t)i * Dm, ln1_b + (size_t)i * Dm, 1.0f);
        gemm(h16b, Dm, inT + (size_t)i * 1024 * 256, in_b + (size_t)i * 1024,
             nullptr, 0, ug, 1024, nullptr, 0, 1024, 256, 0);
        mamba_conv_kernel<<<(M * DId) / 256, 256, 0, stream>>>(
            ug, conv_w + (size_t)i * DId * 4, conv_b + (size_t)i * DId, ucb, uc16);
        gemm(uc16, DId, dtT + (size_t)i * 64 * 512, dt_b + (size_t)i * DSd,
             nullptr, 0, dtb, DSd, nullptr, 0, DSd, DId, 0);
        gemm(uc16, DId, BT + (size_t)i * 64 * 512, nullptr,
             nullptr, 0, btb, DSd, nullptr, 0, DSd, DId, 0);
        gemm(uc16, DId, CT + (size_t)i * 64 * 512, nullptr,
             nullptr, 0, ctb, DSd, nullptr, 0, DSd, DId, 0);
        dt_prep_kernel<<<(M * DSd) / 256, 256, 0, stream>>>(dtb, btb, A_log + (size_t)i * DSd);
        scan_kernel<<<Bq * DSd, 256, 0, stream>>>(dtb, btb, ctb, hC16);
        gemm(hC16, DSd, osT + (size_t)i * 512 * 64, nullptr,
             nullptr, 0, ydi, DId, nullptr, 0, DId, DSd, 0);
        gate_kernel<<<(M * DId) / 256, 256, 0, stream>>>(ydi, ucb, ug,
                                                         D_vec + (size_t)i * DId, y16);
        gemm(y16, DId, outT + (size_t)i * 256 * 512, out_b + (size_t)i * Dm,
             xb, Dm, xb, Dm, nullptr, 0, Dm, DId, 0);

        // ---- Multi-scale conv branch ----
        modulate(ln2_g + (size_t)i * Dm, ln2_b + (size_t)i * Dm, 0.5f);
        for (int j = 0; j < 3; ++j) {
            ms_conv_kernel<<<(M * Dm) / 256, 256, 0, stream>>>(
                hb, dw_w + ((size_t)i * 3 + j) * Dm * 3,
                dw_b + ((size_t)i * 3 + j) * Dm, dils[j], c16);
            gemm(c16, Dm, pwT + ((size_t)i * 3 + j) * 256 * 256,
                 pw_b + (size_t)i * 3 * Dm + (size_t)j * Dm,
                 nullptr, 0, nullptr, 0, concat16 + j * Dm, 768, Dm, Dm, 0);
        }
        gemm(concat16, 768, f1T + (size_t)i * 512 * 768, f1_b + (size_t)i * 512,
             nullptr, 0, nullptr, 0, mid16, 512, 512, 768, 1);
        gemm(mid16, 512, f2T + (size_t)i * 256 * 512, f2_b + (size_t)i * Dm,
             xb, Dm, xb, Dm, nullptr, 0, Dm, 512, 0);

        // ---- FFN branch ----
        modulate(ln3_g + (size_t)i * Dm, ln3_b + (size_t)i * Dm, 0.5f);
        gemm(h16b, Dm, ffn1T + (size_t)i * 1024 * 256, ffn1_b + (size_t)i * 1024,
             nullptr, 0, nullptr, 0, ff16, 1024, 1024, 256, 1);
        gemm(ff16, 1024, ffn2T + (size_t)i * 256 * 1024, ffn2_b + (size_t)i * Dm,
             xb, Dm, xb, Dm, nullptr, 0, Dm, 1024, 0);
    }

    final_ln_kernel<<<M, 256, 0, stream>>>(xb, final_g, final_b, (float*)d_out);
}
